// HierarchicalGraphEncoder_86663850098764
// MI455X (gfx1250) — compile-verified, matrix operand order checked by reading, not run on hardware
//
#include <hip/hip_runtime.h>

#define N_NODES 500000
#define B_GRAPH 16384
#define NDIM    128
#define HID     256
#define SCAF    128
#define FIN     384
#define LN_EPS  1e-5f

typedef unsigned short u16;
typedef __attribute__((ext_vector_type(16))) __bf16 v16bf;
typedef __attribute__((ext_vector_type(8)))  __bf16 v8bf;
typedef __attribute__((ext_vector_type(8)))  float  v8f;

union V16U { v16bf v; v8bf h[2]; };

__device__ __forceinline__ u16 f2bf(float f) {
  union { float f; unsigned u; } x{f};
  unsigned r = x.u + 0x7FFFu + ((x.u >> 16) & 1u);   // round-to-nearest-even
  return (u16)(r >> 16);
}
__device__ __forceinline__ float bf2f(u16 u) {
  union { unsigned u; float f; } x; x.u = ((unsigned)u) << 16; return x.f;
}
// CDNA5 16-bit A/B operand: per lane, elements 0..7 = K granule (half*8),
// elements 8..15 = K granule (16 + half*8). Two 16B chunks -> v16bf.
__device__ __forceinline__ v16bf ld16(const u16* p0, const u16* p1) {
  V16U u;
  u.h[0] = *(const v8bf*)p0;
  u.h[1] = *(const v8bf*)p1;
  return u.v;
}

// ---- 32-row GEMM, K-outer: A loaded once per K-step, reused by 4 col tiles;
//      B loaded once per (K,tile), reused by 2 row tiles. 8 live accumulators.
// A: 32 x K bf16 (lda, LDS). Bt: NOUT x K bf16 (global). out: 32 x ldo f32 LDS.
__device__ __forceinline__ void wave_gemm32(
    const u16* A, int lda, const u16* __restrict__ Bt, int ldb, int ksteps,
    float* out, int ldo, const float* __restrict__ bias, int wave, int lane)
{
  const int r    = lane & 15;
  const int half = (lane >> 4) << 3;     // K-granule select AND D-row offset
  const u16* a0row = A + (size_t)r * lda;
  const u16* a1row = A + (size_t)(16 + r) * lda;
  const int colbase = wave * 64 + r;
  v8f c[2][4] = {};
#pragma unroll
  for (int ks = 0; ks < ksteps; ++ks) {
    const int kb = ks * 32;
    const v16bf a0 = ld16(a0row + kb + half, a0row + kb + 16 + half);
    const v16bf a1 = ld16(a1row + kb + half, a1row + kb + 16 + half);
#pragma unroll
    for (int t = 0; t < 4; ++t) {
      const u16* brow = Bt + (size_t)(colbase + t * 16) * ldb + kb;
      const v16bf b = ld16(brow + half, brow + 16 + half);
      c[0][t] = __builtin_amdgcn_wmma_f32_16x16x32_bf16(false, a0, false, b, (short)0, c[0][t], false, false);
      c[1][t] = __builtin_amdgcn_wmma_f32_16x16x32_bf16(false, a1, false, b, (short)0, c[1][t], false, false);
    }
  }
#pragma unroll
  for (int t = 0; t < 4; ++t) {
    const int col = colbase + t * 16;
    const float bv = bias[col];
#pragma unroll
    for (int q = 0; q < 8; ++q) {
      out[(size_t)(half + q)      * ldo + col] = c[0][t][q] + bv;
      out[(size_t)(16 + half + q) * ldo + col] = c[1][t][q] + bv;
    }
  }
}

// ---- LN for 32 rows, 4 threads/row (tid 0..127) -> bf16 LDS tile -----------
__device__ __forceinline__ void ln32_to_bf16(
    const float* tile, int W, const float* __restrict__ g,
    const float* __restrict__ bt, bool relu, u16* dst, int tid)
{
  const int row = tid >> 2, sub = tid & 3;
  const float* rp = tile + (size_t)row * W;
  float s = 0.f, ss = 0.f;
  for (int i = sub; i < W; i += 4) { float x = rp[i]; s += x; ss += x * x; }
  s  += __shfl_xor(s, 1);  s  += __shfl_xor(s, 2);
  ss += __shfl_xor(ss, 1); ss += __shfl_xor(ss, 2);
  const float m  = s / W;
  const float rs = rsqrtf(ss / W - m * m + LN_EPS);
  for (int i = sub; i < W; i += 4) {
    float y = (rp[i] - m) * rs * g[i] + bt[i];
    if (relu) y = fmaxf(y, 0.f);
    dst[(size_t)row * W + i] = f2bf(y);
  }
}

// ---- LN for 32 rows, in-place f32 (for vectorized f32 store afterwards) ----
__device__ __forceinline__ void ln32_inplace(
    float* tile, int W, const float* __restrict__ g,
    const float* __restrict__ bt, bool relu, int tid)
{
  const int row = tid >> 2, sub = tid & 3;
  float* rp = tile + (size_t)row * W;
  float s = 0.f, ss = 0.f;
  for (int i = sub; i < W; i += 4) { float x = rp[i]; s += x; ss += x * x; }
  s  += __shfl_xor(s, 1);  s  += __shfl_xor(s, 2);
  ss += __shfl_xor(ss, 1); ss += __shfl_xor(ss, 2);
  const float m  = s / W;
  const float rs = rsqrtf(ss / W - m * m + LN_EPS);
  for (int i = sub; i < W; i += 4) {
    float y = (rp[i] - m) * rs * g[i] + bt[i];
    if (relu) y = fmaxf(y, 0.f);
    rp[i] = y;
  }
}

// ---------------- 16-row variant (scaffold path, tiny), K-outer -------------
__device__ __forceinline__ void wave_gemm16(
    const u16* A, int lda, const u16* __restrict__ Bt, int ldb, int ksteps,
    float* out, int ldo, const float* __restrict__ bias,
    int wave, int lane, int tiles_per_wave)
{
  const int r    = lane & 15;
  const int half = (lane >> 4) << 3;
  const u16* arow = A + (size_t)r * lda;
  v8f c[2] = {};
#pragma unroll
  for (int ks = 0; ks < ksteps; ++ks) {
    const int kb = ks * 32;
    const v16bf a = ld16(arow + kb + half, arow + kb + 16 + half);
#pragma unroll
    for (int t = 0; t < 2; ++t) {
      const int col = (wave * tiles_per_wave + t) * 16 + r;
      const u16* brow = Bt + (size_t)col * ldb + kb;
      const v16bf b = ld16(brow + half, brow + 16 + half);
      c[t] = __builtin_amdgcn_wmma_f32_16x16x32_bf16(false, a, false, b, (short)0, c[t], false, false);
    }
  }
#pragma unroll
  for (int t = 0; t < 2; ++t) {
    const int col = (wave * tiles_per_wave + t) * 16 + r;
    const float bv = bias[col];
#pragma unroll
    for (int q = 0; q < 8; ++q)
      out[(size_t)(half + q) * ldo + col] = c[t][q] + bv;
  }
}

__device__ __forceinline__ void ln_rows(
    const float* tile, int W, const float* __restrict__ g,
    const float* __restrict__ bt, bool relu,
    u16* bf_dst, int ld_bf, float* f_dst, size_t ld_f, int tid)
{
  const int row = tid >> 3, sub = tid & 7;
  const float* rp = tile + (size_t)row * W;
  float s = 0.f, ss = 0.f;
  for (int i = sub; i < W; i += 8) { float x = rp[i]; s += x; ss += x * x; }
  s  += __shfl_xor(s, 1);  s  += __shfl_xor(s, 2);  s  += __shfl_xor(s, 4);
  ss += __shfl_xor(ss, 1); ss += __shfl_xor(ss, 2); ss += __shfl_xor(ss, 4);
  const float m  = s / W;
  const float rs = rsqrtf(ss / W - m * m + LN_EPS);
  for (int i = sub; i < W; i += 8) {
    float y = (rp[i] - m) * rs * g[i] + bt[i];
    if (relu) y = fmaxf(y, 0.f);
    if (bf_dst) bf_dst[(size_t)row * ld_bf + i] = f2bf(y);
    if (f_dst)  f_dst[(size_t)row * ld_f + i]  = y;
  }
}

__global__ void k_zero2(float* __restrict__ a, int na, float* __restrict__ b, int nb) {
  int i = blockIdx.x * blockDim.x + threadIdx.x;
  if (i < na) a[i] = 0.f;
  else if (i < na + nb) b[i - na] = 0.f;
}

__global__ void k_transpose_bf(const float* __restrict__ W, u16* __restrict__ WT,
                               int K, int N) {
  int idx = blockIdx.x * blockDim.x + threadIdx.x;
  if (idx >= K * N) return;
  int k = idx / N, n = idx % N;
  WT[(size_t)n * K + k] = f2bf(W[idx]);
}

// ---- node encoder: 32 rows/block; h = relu(LN(relu(LN(X@W0+b0))@W1+b1)) ----
__global__ __launch_bounds__(128) void k_node_enc(
    const float* __restrict__ X,
    const u16* __restrict__ W0T, const float* __restrict__ b0,
    const float* __restrict__ g0, const float* __restrict__ bt0,
    const u16* __restrict__ W1T, const float* __restrict__ b1,
    const float* __restrict__ g1, const float* __restrict__ bt1,
    u16* __restrict__ h_out)
{
  // hbuf serves as the X tile (32x128) for GEMM1, then the h tile (32x256).
  __shared__ __align__(32) u16   hbuf[32 * HID];   // 16 KB
  __shared__ __align__(32) float acc[32 * HID];    // 32 KB
  const int tid = threadIdx.x, wave = tid >> 5, lane = tid & 31;
  const size_t row0 = (size_t)blockIdx.x * 32;

  // stage X tile -> bf16 LDS, vectorized (float4 in, 8B out)
  for (int v = tid; v < 32 * NDIM / 4; v += 128) {
    const int e = v * 4;
    const float4 x = *(const float4*)(X + row0 * NDIM + e);
    uint2 pk;
    pk.x = (unsigned)f2bf(x.x) | ((unsigned)f2bf(x.y) << 16);
    pk.y = (unsigned)f2bf(x.z) | ((unsigned)f2bf(x.w) << 16);
    *(uint2*)&hbuf[e] = pk;
  }
  __syncthreads();
  wave_gemm32(hbuf, NDIM, W0T, NDIM, NDIM / 32, acc, HID, b0, wave, lane);
  __syncthreads();
  ln32_to_bf16(acc, HID, g0, bt0, true, hbuf, tid);   // overwrites X tile (dead)
  __syncthreads();
  wave_gemm32(hbuf, HID, W1T, HID, HID / 32, acc, HID, b1, wave, lane);
  __syncthreads();
  ln32_to_bf16(acc, HID, g1, bt1, true, hbuf, tid);
  __syncthreads();
  // vectorized 16B drain of h tile
  for (int v = tid; v < 32 * HID / 8; v += 128) {
    const int e = v * 8;
    *(uint4*)(h_out + row0 * HID + e) = *(const uint4*)&hbuf[e];
  }
}

// ---- sorted-batch segmented mean-pool (sum + count), boundary atomics ----
__global__ __launch_bounds__(256) void k_segsum(
    const u16* __restrict__ h_bf, const int* __restrict__ batch,
    float* __restrict__ sums, float* __restrict__ cnts)
{
  __shared__ int sb[256];
  const int tid = threadIdx.x;
  const size_t start = (size_t)blockIdx.x * 256;
  int rows = (int)((start + 256 <= (size_t)N_NODES) ? 256 : (size_t)N_NODES - start);
  if (tid < rows) sb[tid] = batch[start + tid];
  __syncthreads();
  const int c = tid;
  float acc = 0.f;
  int seg = sb[0], runstart = 0;
  for (int r = 0; r < rows; ++r) {
    int s = sb[r];
    if (s != seg) {
      atomicAdd(&sums[(size_t)seg * HID + c], acc);
      if (c == 0) atomicAdd(&cnts[seg], (float)(r - runstart));
      acc = 0.f; seg = s; runstart = r;
    }
    acc += bf2f(h_bf[(start + r) * HID + c]);
  }
  atomicAdd(&sums[(size_t)seg * HID + c], acc);
  if (c == 0) atomicAdd(&cnts[seg], (float)(rows - runstart));
}

// ---- scaffold encoder on pooled [B,256] (16 rows/block; tiny workload) ----
__global__ __launch_bounds__(128) void k_scaffold(
    const float* __restrict__ sums, const float* __restrict__ cnts,
    const u16* __restrict__ Ws0T, const float* __restrict__ bs0,
    const float* __restrict__ gs0, const float* __restrict__ bts0,
    const u16* __restrict__ Ws1T, const float* __restrict__ bs1,
    const float* __restrict__ gs1, const float* __restrict__ bts1,
    u16* __restrict__ scaf_bf, float* __restrict__ scaf_out)
{
  __shared__ __align__(32) u16   pb[16 * HID];
  __shared__ __align__(32) float acc[16 * SCAF];
  __shared__ __align__(32) u16   sbuf[16 * SCAF];
  const int tid = threadIdx.x, wave = tid >> 5, lane = tid & 31;
  const size_t row0 = (size_t)blockIdx.x * 16;
  for (int i = tid; i < 16 * HID; i += 128) {
    const int rr = i / HID;
    const float cv = cnts[row0 + rr];
    pb[i] = f2bf(sums[row0 * HID + i] / fmaxf(cv, 1.0f));
  }
  __syncthreads();
  wave_gemm16(pb, HID, Ws0T, HID, HID / 32, acc, SCAF, bs0, wave, lane, 2);
  __syncthreads();
  ln_rows(acc, SCAF, gs0, bts0, true, sbuf, SCAF, nullptr, 0, tid);
  __syncthreads();
  wave_gemm16(sbuf, SCAF, Ws1T, SCAF, SCAF / 32, acc, SCAF, bs1, wave, lane, 2);
  __syncthreads();
  ln_rows(acc, SCAF, gs1, bts1, false, scaf_bf + row0 * SCAF, SCAF,
          scaf_out + row0 * SCAF, SCAF, tid);
}

// ---- fuse: 32 rows/block; relu(LN([h | scaffold[batch]] @ Wf + bf)) ----
// A operands staged in LDS once per block; K-outer GEMM over both segments.
__global__ __launch_bounds__(128) void k_fuse(
    const u16* __restrict__ h_bf, const u16* __restrict__ scafbf,
    const int* __restrict__ batch, const u16* __restrict__ WfT,
    const float* __restrict__ bias, const float* __restrict__ g,
    const float* __restrict__ bt, float* __restrict__ out)
{
  __shared__ __align__(32) u16   ht[32 * HID];     // 16 KB
  __shared__ __align__(32) u16   st[32 * SCAF];    //  8 KB
  __shared__ __align__(32) float acc[32 * HID];    // 32 KB
  __shared__ int sseg[32];
  const int tid = threadIdx.x, wave = tid >> 5, lane = tid & 31;
  const size_t row0 = (size_t)blockIdx.x * 32;
  if (tid < 32) sseg[tid] = batch[row0 + tid];
  __syncthreads();
  // stage h tile (coalesced) and gathered scaffold rows
  for (int v = tid; v < 32 * HID / 8; v += 128) {
    const int e = v * 8;
    *(uint4*)&ht[e] = *(const uint4*)(h_bf + row0 * HID + e);
  }
  for (int v = tid; v < 32 * SCAF / 8; v += 128) {
    const int e = v * 8;
    const int row = e >> 7, cidx = e & (SCAF - 1);
    *(uint4*)&st[e] = *(const uint4*)(scafbf + (size_t)sseg[row] * SCAF + cidx);
  }
  __syncthreads();

  const int r    = lane & 15;
  const int half = (lane >> 4) << 3;
  const u16* h0 = ht + (size_t)r * HID;
  const u16* h1 = ht + (size_t)(16 + r) * HID;
  const u16* s0 = st + (size_t)r * SCAF;
  const u16* s1 = st + (size_t)(16 + r) * SCAF;
  const int colbase = wave * 64 + r;
  v8f c[2][4] = {};
#pragma unroll
  for (int ks = 0; ks < HID / 32; ++ks) {          // K = 0..255 : h part
    const int kb = ks * 32;
    const v16bf a0 = ld16(h0 + kb + half, h0 + kb + 16 + half);
    const v16bf a1 = ld16(h1 + kb + half, h1 + kb + 16 + half);
#pragma unroll
    for (int t = 0; t < 4; ++t) {
      const u16* brow = WfT + (size_t)(colbase + t * 16) * FIN + kb;
      const v16bf b = ld16(brow + half, brow + 16 + half);
      c[0][t] = __builtin_amdgcn_wmma_f32_16x16x32_bf16(false, a0, false, b, (short)0, c[0][t], false, false);
      c[1][t] = __builtin_amdgcn_wmma_f32_16x16x32_bf16(false, a1, false, b, (short)0, c[1][t], false, false);
    }
  }
#pragma unroll
  for (int ks = 0; ks < SCAF / 32; ++ks) {         // K = 256..383 : scaffold
    const int kb = ks * 32;
    const v16bf a0 = ld16(s0 + kb + half, s0 + kb + 16 + half);
    const v16bf a1 = ld16(s1 + kb + half, s1 + kb + 16 + half);
#pragma unroll
    for (int t = 0; t < 4; ++t) {
      const u16* brow = WfT + (size_t)(colbase + t * 16) * FIN + HID + kb;
      const v16bf b = ld16(brow + half, brow + 16 + half);
      c[0][t] = __builtin_amdgcn_wmma_f32_16x16x32_bf16(false, a0, false, b, (short)0, c[0][t], false, false);
      c[1][t] = __builtin_amdgcn_wmma_f32_16x16x32_bf16(false, a1, false, b, (short)0, c[1][t], false, false);
    }
  }
#pragma unroll
  for (int t = 0; t < 4; ++t) {
    const int col = colbase + t * 16;
    const float bv = bias[col];
#pragma unroll
    for (int q = 0; q < 8; ++q) {
      acc[(size_t)(half + q)      * HID + col] = c[0][t][q] + bv;
      acc[(size_t)(16 + half + q) * HID + col] = c[1][t][q] + bv;
    }
  }
  __syncthreads();
  ln32_inplace(acc, HID, g, bt, true, tid);
  __syncthreads();
  // vectorized 16B drain of the fused tile (largest HBM stream)
  for (int v = tid; v < 32 * HID / 4; v += 128) {
    const int e = v * 4;
    *(float4*)(out + row0 * HID + e) = *(const float4*)&acc[e];
  }
}

extern "C" void kernel_launch(void* const* d_in, const int* in_sizes, int n_in,
                              void* d_out, int out_size, void* d_ws, size_t ws_size,
                              hipStream_t stream) {
  const float* X    = (const float*)d_in[0];
  const int*   bat  = (const int*)  d_in[2];
  const float* W0   = (const float*)d_in[3];
  const float* b0   = (const float*)d_in[4];
  const float* g0   = (const float*)d_in[5];
  const float* bt0  = (const float*)d_in[6];
  const float* W1   = (const float*)d_in[7];
  const float* b1   = (const float*)d_in[8];
  const float* g1   = (const float*)d_in[9];
  const float* bt1  = (const float*)d_in[10];
  const float* Ws0  = (const float*)d_in[11];
  const float* bs0  = (const float*)d_in[12];
  const float* gs0  = (const float*)d_in[13];
  const float* bts0 = (const float*)d_in[14];
  const float* Ws1  = (const float*)d_in[15];
  const float* bs1  = (const float*)d_in[16];
  const float* gs1  = (const float*)d_in[17];
  const float* bts1 = (const float*)d_in[18];
  const float* Wf   = (const float*)d_in[19];
  const float* bfv  = (const float*)d_in[20];
  const float* gf   = (const float*)d_in[21];
  const float* btf  = (const float*)d_in[22];

  float* out_fused = (float*)d_out;
  float* out_scaf  = (float*)d_out + (size_t)N_NODES * HID;

  char* ws = (char*)d_ws;
  size_t off = 0;
  auto take = [&](size_t bytes) -> char* {
    char* p = ws + off;
    off = (off + bytes + 255) & ~(size_t)255;
    return p;
  };
  u16*   h_bf   = (u16*)  take((size_t)N_NODES * HID * 2);
  u16*   scafbf = (u16*)  take((size_t)B_GRAPH * SCAF * 2);
  float* sums   = (float*)take((size_t)B_GRAPH * HID * 4);
  float* cnts   = (float*)take((size_t)B_GRAPH * 4);
  u16*   W0T    = (u16*)  take((size_t)HID  * NDIM * 2);
  u16*   W1T    = (u16*)  take((size_t)HID  * HID  * 2);
  u16*   Ws0T   = (u16*)  take((size_t)SCAF * HID  * 2);
  u16*   Ws1T   = (u16*)  take((size_t)SCAF * SCAF * 2);
  u16*   WfT    = (u16*)  take((size_t)HID  * FIN  * 2);

  // prep: zero pooling accumulators (atomics!) and bf16-transpose weights
  {
    int nz = B_GRAPH * HID + B_GRAPH;
    k_zero2<<<(nz + 255) / 256, 256, 0, stream>>>(sums, B_GRAPH * HID, cnts, B_GRAPH);
    k_transpose_bf<<<(NDIM * HID + 255) / 256, 256, 0, stream>>>(W0,  W0T,  NDIM, HID);
    k_transpose_bf<<<(HID  * HID + 255) / 256, 256, 0, stream>>>(W1,  W1T,  HID,  HID);
    k_transpose_bf<<<(HID  * SCAF + 255) / 256, 256, 0, stream>>>(Ws0, Ws0T, HID,  SCAF);
    k_transpose_bf<<<(SCAF * SCAF + 255) / 256, 256, 0, stream>>>(Ws1, Ws1T, SCAF, SCAF);
    k_transpose_bf<<<(FIN  * HID + 255) / 256, 256, 0, stream>>>(Wf,  WfT,  FIN,  HID);
  }

  k_node_enc<<<N_NODES / 32, 128, 0, stream>>>(X, W0T, b0, g0, bt0,
                                               W1T, b1, g1, bt1, h_bf);
  k_segsum<<<(N_NODES + 255) / 256, 256, 0, stream>>>(h_bf, bat, sums, cnts);
  k_scaffold<<<B_GRAPH / 16, 128, 0, stream>>>(sums, cnts, Ws0T, bs0, gs0, bts0,
                                               Ws1T, bs1, gs1, bts1, scafbf, out_scaf);
  k_fuse<<<N_NODES / 32, 128, 0, stream>>>(h_bf, scafbf, bat, WfT,
                                           bfv, gf, btf, out_fused);
}
